// MultiHeadAttention_5712306504268
// MI455X (gfx1250) — compile-verified, matrix operand order checked by reading, not run on hardware
//
#include <hip/hip_runtime.h>
#include <hip/hip_bf16.h>

typedef __attribute__((ext_vector_type(16))) _Float16 v16h;
typedef __attribute__((ext_vector_type(8)))  _Float16 v8h;
typedef __attribute__((ext_vector_type(8)))  float    v8f;
typedef __attribute__((ext_vector_type(4)))  float    v4f;

#define BM 128
#define BN 128
#define BK 32
#define LDSS 40   // f16 elements per LDS row: 80B, multiple of 16B so v8h loads stay aligned

// ---------------------------------------------------------------------------
// Generic WMMA GEMM:  C = scale * (A @ B) + bias + add
//   A: fp32 [M,K] row-major (lda); M is an exact multiple of BM (grid-sized).
//   B: BT=false -> fp32 [K,N] row-major (ldb);  BT=true -> fp32 [N,K] (ldb)
//   K is an exact multiple of BK and K/BK >= 2. N may be < BN (branch-free
//   clamp+mask on B staging, exec-guard only on the C store).
//   fp32 operands convert to f16 while staging into LDS; f32 accumulate via
//   v_wmma_f32_16x16x32_f16. 256 threads = 8 waves; wave grid 2(M)x4(N); each
//   wave computes 4x2 = 8 16x16 tiles => block tile 128x128.
//   Double-buffered LDS: tile t+1's global loads are in flight during tile
//   t's WMMAs; one barrier per K-step.
// ---------------------------------------------------------------------------
template <bool BT>
__global__ __launch_bounds__(256)
void gemm_wmma(const float* __restrict__ A, const float* __restrict__ Bm,
               const float* __restrict__ bias, const float* __restrict__ add,
               float* __restrict__ C,
               int M, int N, int K, int lda, int ldb, int ldc,
               long long strideA, long long strideB, long long strideC,
               float scale)
{
    __shared__ _Float16 As[2][BM * LDSS];
    __shared__ _Float16 Bs[2][BN * LDSS];

    const int b = blockIdx.z;
    A  += (long long)b * strideA;
    Bm += (long long)b * strideB;
    C  += (long long)b * strideC;
    const float* addp = add ? add + (long long)b * strideC : nullptr;

    const int tid   = threadIdx.x;
    const int lane  = tid & 31;
    const int wave  = tid >> 5;     // 0..7
    const int waveM = wave >> 2;    // 0..1 -> 64 rows each
    const int waveN = wave & 3;     // 0..3 -> 32 cols each

    const int blockRow = blockIdx.y * BM;
    const int blockCol = blockIdx.x * BN;

    const int l16     = lane & 15;
    const int halfSel = lane >> 4;        // 0: lanes 0-15, 1: lanes 16-31
    const int c0      = halfSel << 3;     // K sub-offset per ISA A/B 16-bit layout

    // ---- staging coordinates (fixed per thread; +32 rows / +8 k-rows per it)
    const int rowS = tid >> 3;            // 0..31   (A rows / BT B rows)
    const int colS = (tid & 7) << 2;      // 0..28   (k offset, vec4)
    const int kkB  = tid >> 5;            // 0..7    (non-BT B k-row)
    const int nvB  = lane << 2;           // 0..124  (non-BT B n offset, vec4)

    const float* aBase = A + (long long)(blockRow + rowS) * lda + colS;

    const v4f vzero = {0.f, 0.f, 0.f, 0.f};

    v8f acc[4][2];
    {
        v8f z = {0.f, 0.f, 0.f, 0.f, 0.f, 0.f, 0.f, 0.f};
        #pragma unroll
        for (int i = 0; i < 4; ++i)
            #pragma unroll
            for (int j = 0; j < 2; ++j)
                acc[i][j] = z;
    }

    // ---- helpers -----------------------------------------------------------
    auto loadTile = [&](int k0, v4f (&ar)[4], v4f (&br)[4]) {
        #pragma unroll
        for (int it = 0; it < 4; ++it)
            ar[it] = *(const v4f*)(aBase + k0 + (long long)(it * 32) * lda);
        if constexpr (BT) {
            #pragma unroll
            for (int it = 0; it < 4; ++it) {
                int gn  = blockCol + rowS + it * 32;
                bool ok = gn < N;
                v4f d = *(const v4f*)(Bm + (long long)(ok ? gn : 0) * ldb + k0 + colS);
                br[it] = ok ? d : vzero;
            }
        } else {
            int gn  = blockCol + nvB;
            bool ok = gn < N;
            const float* bp = Bm + (ok ? gn : 0) + k0 * (long long)ldb;
            #pragma unroll
            for (int it = 0; it < 4; ++it) {
                v4f d = *(const v4f*)(bp + (long long)(kkB + it * 8) * ldb);
                br[it] = ok ? d : vzero;
            }
        }
    };

    auto storeTile = [&](int buf, const v4f (&ar)[4], const v4f (&br)[4]) {
        #pragma unroll
        for (int it = 0; it < 4; ++it) {
            _Float16* dst = &As[buf][(rowS + it * 32) * LDSS + colS];
            dst[0] = (_Float16)ar[it].x; dst[1] = (_Float16)ar[it].y;
            dst[2] = (_Float16)ar[it].z; dst[3] = (_Float16)ar[it].w;
        }
        if constexpr (BT) {
            #pragma unroll
            for (int it = 0; it < 4; ++it) {
                _Float16* dst = &Bs[buf][(rowS + it * 32) * LDSS + colS];
                dst[0] = (_Float16)br[it].x; dst[1] = (_Float16)br[it].y;
                dst[2] = (_Float16)br[it].z; dst[3] = (_Float16)br[it].w;
            }
        } else {
            #pragma unroll
            for (int it = 0; it < 4; ++it) {
                int kk = kkB + it * 8;
                Bs[buf][(nvB + 0) * LDSS + kk] = (_Float16)br[it].x;
                Bs[buf][(nvB + 1) * LDSS + kk] = (_Float16)br[it].y;
                Bs[buf][(nvB + 2) * LDSS + kk] = (_Float16)br[it].z;
                Bs[buf][(nvB + 3) * LDSS + kk] = (_Float16)br[it].w;
            }
        }
    };
    // ------------------------------------------------------------------------

    const int ntiles = K / BK;

    {   // prologue: stage tile 0
        v4f ar[4], br[4];
        loadTile(0, ar, br);
        storeTile(0, ar, br);
    }

    int cur = 0;
    for (int t = 0; t < ntiles; ++t) {
        __syncthreads();

        // issue next tile's global loads; they stay in flight during the WMMAs
        v4f an[4], bn[4];
        const bool more = (t + 1) < ntiles;
        if (more) loadTile((t + 1) * BK, an, bn);
        if ((t + 2) < ntiles)
            __builtin_prefetch(aBase + (t + 2) * BK, 0, 3);   // global_prefetch_b8

        // ---- fragments per ISA 16-bit operand layout, then 8 WMMAs
        const _Float16* asb = As[cur];
        const _Float16* bsb = Bs[cur];
        v16h afrag[4], bfrag[2];
        #pragma unroll
        for (int mt = 0; mt < 4; ++mt) {
            int r = waveM * 64 + mt * 16 + l16;
            v8h lo = *(const v8h*)&asb[r * LDSS + c0];
            v8h hi = *(const v8h*)&asb[r * LDSS + c0 + 16];
            afrag[mt] = __builtin_shufflevector(lo, hi, 0,1,2,3,4,5,6,7,8,9,10,11,12,13,14,15);
        }
        #pragma unroll
        for (int nt = 0; nt < 2; ++nt) {
            int c = waveN * 32 + nt * 16 + l16;
            v8h lo = *(const v8h*)&bsb[c * LDSS + c0];
            v8h hi = *(const v8h*)&bsb[c * LDSS + c0 + 16];
            bfrag[nt] = __builtin_shufflevector(lo, hi, 0,1,2,3,4,5,6,7,8,9,10,11,12,13,14,15);
        }
        #pragma unroll
        for (int mt = 0; mt < 4; ++mt)
            #pragma unroll
            for (int nt = 0; nt < 2; ++nt)
                acc[mt][nt] = __builtin_amdgcn_wmma_f32_16x16x32_f16(
                    false, afrag[mt], false, bfrag[nt],
                    (short)0, acc[mt][nt], false, false);

        // convert + store next tile into the alternate buffer (no race: other
        // waves only read buffer `cur` this iteration)
        if (more) storeTile(cur ^ 1, an, bn);
        cur ^= 1;
    }

    // ---- epilogue: C/D layout -> lane(l16)=col, VGPR e + halfSel*8 = row
    #pragma unroll
    for (int mt = 0; mt < 4; ++mt) {
        int rowBase = blockRow + waveM * 64 + mt * 16 + (halfSel << 3);
        #pragma unroll
        for (int nt = 0; nt < 2; ++nt) {
            int col = blockCol + waveN * 32 + nt * 16 + l16;
            if (col < N) {
                float bv = bias ? bias[col] : 0.f;
                #pragma unroll
                for (int e = 0; e < 8; ++e) {
                    int row = rowBase + e;                 // M is a multiple of 128 here
                    long long idx = (long long)row * ldc + col;
                    float val = acc[mt][nt][e] * scale + bv;
                    if (addp) val += addp[idx];
                    C[idx] = val;
                }
            }
        }
    }
}

// ---------------------------------------------------------------------------
// Wave-per-row softmax over rows of length 2048 (in place). Block = 8 waves.
// ---------------------------------------------------------------------------
__global__ __launch_bounds__(256)
void softmax_rows(float* __restrict__ att)
{
    const int row  = blockIdx.x * 8 + (threadIdx.x >> 5);
    const int lane = threadIdx.x & 31;
    float* p = att + (long long)row * 2048;

    float x[64];
    float m = -3.0e38f;
    #pragma unroll
    for (int i = 0; i < 16; ++i) {
        v4f d = *(const v4f*)(p + (((i << 5) + lane) << 2));
        x[4 * i + 0] = d.x; x[4 * i + 1] = d.y; x[4 * i + 2] = d.z; x[4 * i + 3] = d.w;
        m = fmaxf(m, fmaxf(fmaxf(d.x, d.y), fmaxf(d.z, d.w)));
    }
    #pragma unroll
    for (int off = 16; off > 0; off >>= 1)
        m = fmaxf(m, __shfl_xor(m, off, 32));

    float s = 0.f;
    #pragma unroll
    for (int i = 0; i < 64; ++i) { x[i] = __expf(x[i] - m); s += x[i]; }
    #pragma unroll
    for (int off = 16; off > 0; off >>= 1)
        s += __shfl_xor(s, off, 32);
    float inv = 1.f / s;

    #pragma unroll
    for (int i = 0; i < 16; ++i) {
        v4f d;
        d.x = x[4 * i + 0] * inv; d.y = x[4 * i + 1] * inv;
        d.z = x[4 * i + 2] * inv; d.w = x[4 * i + 3] * inv;
        *(v4f*)(p + (((i << 5) + lane) << 2)) = d;
    }
}

// ---------------------------------------------------------------------------
// LayerNorm over rows of 1024; one block per row, 4 elements/thread.
// ---------------------------------------------------------------------------
__global__ __launch_bounds__(256)
void layernorm_rows(const float* __restrict__ x, const float* __restrict__ gamma,
                    const float* __restrict__ beta, float* __restrict__ out)
{
    const int row  = blockIdx.x;
    const int tid  = threadIdx.x;
    const int lane = tid & 31;
    const int wave = tid >> 5;

    const float* p = x + (long long)row * 1024;
    v4f d = *(const v4f*)(p + (tid << 2));

    float s  = d.x + d.y + d.z + d.w;
    float s2 = d.x * d.x + d.y * d.y + d.z * d.z + d.w * d.w;
    #pragma unroll
    for (int off = 16; off > 0; off >>= 1) {
        s  += __shfl_xor(s,  off, 32);
        s2 += __shfl_xor(s2, off, 32);
    }
    __shared__ float rs[8], rs2[8];
    if (lane == 0) { rs[wave] = s; rs2[wave] = s2; }
    __syncthreads();
    float ts = 0.f, ts2 = 0.f;
    #pragma unroll
    for (int i = 0; i < 8; ++i) { ts += rs[i]; ts2 += rs2[i]; }

    const float mu  = ts * (1.f / 1024.f);
    const float var = ts2 * (1.f / 1024.f) - mu * mu;
    const float r   = rsqrtf(var + 1e-5f);

    v4f g = *(const v4f*)(gamma + (tid << 2));
    v4f bb = *(const v4f*)(beta + (tid << 2));
    v4f o;
    o.x = (d.x - mu) * r * g.x + bb.x;
    o.y = (d.y - mu) * r * g.y + bb.y;
    o.z = (d.z - mu) * r * g.z + bb.z;
    o.w = (d.w - mu) * r * g.w + bb.w;
    *(v4f*)(out + (long long)row * 1024 + (tid << 2)) = o;
}

// ---------------------------------------------------------------------------
extern "C" void kernel_launch(void* const* d_in, const int* in_sizes, int n_in,
                              void* d_out, int out_size, void* d_ws, size_t ws_size,
                              hipStream_t stream)
{
    (void)in_sizes; (void)n_in; (void)out_size; (void)ws_size;

    const float* k    = (const float*)d_in[0];
    const float* v    = (const float*)d_in[1];
    const float* q    = (const float*)d_in[2];
    const float* Wq   = (const float*)d_in[3];
    const float* bq   = (const float*)d_in[4];
    const float* Wk   = (const float*)d_in[5];
    const float* bk   = (const float*)d_in[6];
    const float* Wv   = (const float*)d_in[7];
    const float* bv   = (const float*)d_in[8];
    const float* Wres = (const float*)d_in[9];
    const float* bres = (const float*)d_in[10];
    const float* Wf   = (const float*)d_in[11];
    const float* bf   = (const float*)d_in[12];
    const float* gamma= (const float*)d_in[13];
    const float* beta = (const float*)d_in[14];

    const long long BS = 8192;           // B*S rows
    const long long MD = 1024;           // model dim
    float* out  = (float*)d_out;                       // [8192,1024]
    float* attn = out + BS * MD;                       // [64,2048,2048]

    float* ws       = (float*)d_ws;
    float* residual = ws;                              // 8M f32
    float* qh       = ws + 1 * BS * MD;                // 8M f32
    float* kh       = ws + 2 * BS * MD;                // 8M f32
    float* vh       = ws + 3 * BS * MD;                // 8M f32
    float* ctx      = qh;                              // reuse (qh dead after scores)
    float* xbuf     = kh;                              // reuse (kh dead after scores)

    dim3 blk(256);
    dim3 gLin(1024 / BN, 8192 / BM, 1);                // (8, 64)

    // 1-4: input projections, f32 -> WMMA f16 GEMM, bias fused
    gemm_wmma<false><<<gLin, blk, 0, stream>>>(q, Wres, bres, nullptr, residual,
        8192, 1024, 1024, 1024, 1024, 1024, 0, 0, 0, 1.f);
    gemm_wmma<false><<<gLin, blk, 0, stream>>>(q, Wq, bq, nullptr, qh,
        8192, 1024, 1024, 1024, 1024, 1024, 0, 0, 0, 1.f);
    gemm_wmma<false><<<gLin, blk, 0, stream>>>(k, Wk, bk, nullptr, kh,
        8192, 1024, 1024, 1024, 1024, 1024, 0, 0, 0, 1.f);
    gemm_wmma<false><<<gLin, blk, 0, stream>>>(v, Wv, bv, nullptr, vh,
        8192, 1024, 1024, 1024, 1024, 1024, 0, 0, 0, 1.f);

    // 5: scores = scale * qh @ kh^T  (64 batches, K = 64), written to attn output
    dim3 gScore(2048 / BN, 2048 / BM, 64);
    gemm_wmma<true><<<gScore, blk, 0, stream>>>(qh, kh, nullptr, nullptr, attn,
        2048, 2048, 64, 64, 64, 2048,
        2048LL * 64, 2048LL * 64, 2048LL * 2048, 0.125f);

    // 6: row softmax in place on attn
    softmax_rows<<<(64 * 2048) / 8, blk, 0, stream>>>(attn);

    // 7: ctx = attn @ vh  (64 batches, N = 64)
    dim3 gCtx(1, 2048 / BM, 64);
    gemm_wmma<false><<<gCtx, blk, 0, stream>>>(attn, vh, nullptr, nullptr, ctx,
        2048, 64, 2048, 2048, 64, 64,
        2048LL * 2048, 2048LL * 64, 2048LL * 64, 1.f);

    // 8: x = ctx @ Wf + bf + residual (residual add fused in epilogue)
    gemm_wmma<false><<<gLin, blk, 0, stream>>>(ctx, Wf, bf, residual, xbuf,
        8192, 1024, 1024, 1024, 1024, 1024, 0, 0, 0, 1.f);

    // 9: layernorm -> output
    layernorm_rows<<<8192, blk, 0, stream>>>(xbuf, gamma, beta, out);
}
